// ReliabilityTrustMetric_6158983102739
// MI455X (gfx1250) — 8-bit matrix operand order repaired and checked by reading, compiled, not run on hardware
//
#include <hip/hip_runtime.h>
#include <hip/hip_bf16.h>
#include <math.h>

#define BQ 128
#define NN 4096
#define WW 5
#define MM 50
#define BN (BQ * NN)
#define NBLK 64

typedef __attribute__((ext_vector_type(8))) int v8i;
union V8 { v8i v; int s[8]; };

#define OFF_MASKT ((size_t)0)
#define OFF_A8    (OFF_MASKT + (size_t)NN * NN)
#define OFF_PSUM  (OFF_A8 + (size_t)BQ * NN)
#define OFF_DEG   (OFF_PSUM + (size_t)BQ * NN * 4)

__global__ void __launch_bounds__(256)
prep_mask_kernel(const int* __restrict__ adj, unsigned char* __restrict__ maskT) {
    __shared__ unsigned char tile[32][33];
    const int i0 = blockIdx.x * 32;
    const int j0 = blockIdx.y * 32;
    const int tx = threadIdx.x;
    const int ty = threadIdx.y;
#pragma unroll
    for (int r = 0; r < 4; ++r) {
        const int jl = ty + r * 8;
        const int a = __builtin_nontemporal_load(&adj[(size_t)(j0 + jl) * NN + i0 + tx]);
        tile[jl][tx] = (a > 0) ? 1 : 0;
    }
    __syncthreads();
#pragma unroll
    for (int r = 0; r < 4; ++r) {
        const int il = ty + r * 8;
        maskT[(size_t)(i0 + il) * NN + j0 + tx] = tile[tx][il];
    }
}

__global__ void __launch_bounds__(256)
deg_kernel(const unsigned char* __restrict__ maskT, float* __restrict__ deg) {
    __shared__ int red[256];
    const int row = blockIdx.x;
    const int* rowp = (const int*)(maskT + (size_t)row * NN);
    const int t = threadIdx.x;
    int sum = 0;
#pragma unroll
    for (int r = 0; r < NN / 4 / 256; ++r) {
        const int w = rowp[t + r * 256];
        sum += (w & 0xFF) + ((w >> 8) & 0xFF) + ((w >> 16) & 0xFF) + ((w >> 24) & 0xFF);
    }
    red[t] = sum;
    __syncthreads();
    for (int s = 128; s > 0; s >>= 1) {
        if (t < s) red[t] += red[t + s];
        __syncthreads();
    }
    if (t == 0) deg[row] = (float)red[0];
}

__global__ void __launch_bounds__(256)
pf_kernel(const float* __restrict__ fp, unsigned char* __restrict__ A8) {
    const int idx = blockIdx.x * 256 + threadIdx.x;
    A8[idx] = (fp[idx] > 0.5f) ? 1 : 0;
}

__global__ void __launch_bounds__(256)
gemm_kernel(const unsigned char* __restrict__ A8,
            const unsigned char* __restrict__ maskT,
            int* __restrict__ psum) {
    __shared__ unsigned char ldsA[BQ * 64];
    __shared__ unsigned char ldsB[NBLK * 64];

    const int tid  = threadIdx.x;
    const int lane = tid & 31;
    const int wave = tid >> 5;
    const int n0   = blockIdx.x * NBLK;
    const int m0   = wave * 16;

    V8 acc0, acc1, acc2, acc3;
#pragma unroll
    for (int v = 0; v < 8; ++v) {
        acc0.s[v] = 0; acc1.s[v] = 0; acc2.s[v] = 0; acc3.s[v] = 0;
    }

    const int mA  = lane & 15;
    const int khA = (lane >> 4) * 8;
    const int nB  = lane & 15;
    const int khB = (lane >> 4) * 16;

    const int rowA0 = (tid) >> 2;
    const int colA0 = (tid & 3) * 16;

    for (int kk = 0; kk < NN; kk += 64) {
#pragma unroll
        for (int r = 0; r < 2; ++r) {
            const int row = rowA0 + r * 64;
            *(int4*)&ldsA[row * 64 + colA0] =
                *(const int4*)&A8[(size_t)row * NN + kk + colA0];
        }
        *(int4*)&ldsB[rowA0 * 64 + colA0] =
            *(const int4*)&maskT[(size_t)(n0 + rowA0) * NN + kk + colA0];
        if (kk + 64 < NN) {
            __builtin_prefetch(&maskT[(size_t)n0 * NN + kk + 64], 0, 1);
            __builtin_prefetch(&A8[kk + 64], 0, 1);
        }
        __syncthreads();

        V8 aF, bF0, bF1, bF2, bF3;
#pragma unroll
        for (int v = 0; v < 8; ++v) {
            const int ka = (v >> 2) * 32 + ((v & 3) * 4) + khB;
            aF.s[v] = *(const int*)&ldsA[(m0 + mA) * 64 + ka];
            const int kb = (v >> 2) * 32 + ((v & 3) * 4) + khB;
            bF0.s[v] = *(const int*)&ldsB[(nB)      * 64 + kb];
            bF1.s[v] = *(const int*)&ldsB[(16 + nB) * 64 + kb];
            bF2.s[v] = *(const int*)&ldsB[(32 + nB) * 64 + kb];
            bF3.s[v] = *(const int*)&ldsB[(48 + nB) * 64 + kb];
        }
        acc0.v = __builtin_amdgcn_wmma_i32_16x16x64_iu8(false, aF.v, false, bF0.v, acc0.v, true,  false);
        acc1.v = __builtin_amdgcn_wmma_i32_16x16x64_iu8(false, aF.v, false, bF1.v, acc1.v, true,  false);
        acc2.v = __builtin_amdgcn_wmma_i32_16x16x64_iu8(false, aF.v, false, bF2.v, acc2.v, true,  false);
        acc3.v = __builtin_amdgcn_wmma_i32_16x16x64_iu8(false, aF.v, false, bF3.v, acc3.v, false, false);
        __syncthreads();
    }

    const int mrow = m0 + (lane >> 4) * 8;
#pragma unroll
    for (int v = 0; v < 8; ++v) {
        psum[(size_t)(mrow + v) * NN + n0 + nB]      = acc0.s[v];
        psum[(size_t)(mrow + v) * NN + n0 + 16 + nB] = acc1.s[v];
        psum[(size_t)(mrow + v) * NN + n0 + 32 + nB] = acc2.s[v];
        psum[(size_t)(mrow + v) * NN + n0 + 48 + nB] = acc3.s[v];
    }
}

__global__ void __launch_bounds__(256)
finalize_kernel(const float* __restrict__ fault_probs,
                const float* __restrict__ qos,
                const float* __restrict__ hist,
                const float* __restrict__ dropout_u,
                const float* __restrict__ g1p, const float* __restrict__ g2p,
                const float* __restrict__ g3p,
                const float* __restrict__ bq, const float* __restrict__ bs,
                const int* __restrict__ psum,
                const float* __restrict__ deg,
                float* __restrict__ out) {
    const int idx = blockIdx.x * 256 + threadIdx.x;
    const int n = idx & (NN - 1);

    const float p = fault_probs[idx];

    const float d = deg[n];
    const float pmean = (float)psum[idx] / fmaxf(d, 1.0f);
    const float cons = (d == 0.0f) ? 1.0f : ((pmean <= p + 0.3f) ? 1.0f : 0.0f);

    float ss = 0.0f;
#pragma unroll
    for (int c = 0; c < 4; ++c) {
        const float v = (qos[(size_t)idx * 4 + c] - bq[c]) / (bs[c] + 1e-8f);
        ss += v * v;
    }
    const float x = sqrtf(ss);
    const float supp = 1.0f / (1.0f + __expf(-x));

    float h[WW];
    float mean = 0.0f;
#pragma unroll
    for (int w = 0; w < WW; ++w) { h[w] = hist[(size_t)idx * WW + w]; mean += h[w]; }
    mean *= (1.0f / WW);
    float var = 0.0f;
#pragma unroll
    for (int w = 0; w < WW; ++w) { const float dd = h[w] - mean; var += dd * dd; }
    var *= (1.0f / WW);
    const float stab = 1.0f / (1.0f + var);

    float s1 = 0.0f, s2 = 0.0f;
    for (int m = 0; m < MM; ++m) {
        const float u = __builtin_nontemporal_load(&dropout_u[(size_t)m * BN + idx]);
        const float keep = (u >= 0.3f) ? (1.0f / 0.7f) : 0.0f;
        s1 += keep;
        s2 += keep * keep;
    }
    const float mk = s1 * (1.0f / MM);
    const float vk = s2 * (1.0f / MM) - mk * mk;
    const float unc = sqrtf(p * p * vk + 1e-8f);

    const float g1 = g1p[0], g2 = g2p[0], g3 = g3p[0];
    const float gs = g1 + g2 + g3 + 1e-8f;
    const float trust = (g1 / gs) * cons + (g2 / gs) * supp + (g3 / gs) * stab - 0.1f * unc;

    out[idx]            = trust;
    out[BN + idx]       = cons;
    out[2 * BN + idx]   = supp;
    out[3 * BN + idx]   = stab;
    out[4 * BN + idx]   = unc;
}

extern "C" void kernel_launch(void* const* d_in, const int* in_sizes, int n_in,
                              void* d_out, int out_size, void* d_ws, size_t ws_size,
                              hipStream_t stream) {
    const float* fault_probs = (const float*)d_in[0];
    const float* qos         = (const float*)d_in[1];
    const float* hist        = (const float*)d_in[2];
    const int*   adj         = (const int*)d_in[3];
    const float* dropout_u   = (const float*)d_in[4];
    const float* g1          = (const float*)d_in[5];
    const float* g2          = (const float*)d_in[6];
    const float* g3          = (const float*)d_in[7];
    const float* bq          = (const float*)d_in[8];
    const float* bs          = (const float*)d_in[9];

    unsigned char* ws    = (unsigned char*)d_ws;
    unsigned char* maskT = ws + OFF_MASKT;
    unsigned char* A8    = ws + OFF_A8;
    int*   psum          = (int*)(ws + OFF_PSUM);
    float* deg           = (float*)(ws + OFF_DEG);

    dim3 tb(32, 8);
    dim3 tg(NN / 32, NN / 32);
    prep_mask_kernel<<<tg, tb, 0, stream>>>(adj, maskT);
    deg_kernel<<<NN, 256, 0, stream>>>(maskT, deg);
    pf_kernel<<<BN / 256, 256, 0, stream>>>(fault_probs, A8);
    gemm_kernel<<<NN / NBLK, 256, 0, stream>>>(A8, maskT, psum);
    finalize_kernel<<<BN / 256, 256, 0, stream>>>(fault_probs, qos, hist, dropout_u,
                                                  g1, g2, g3, bq, bs, psum, deg,
                                                  (float*)d_out);
}
